// CausalSelfAttention_18356690223660
// MI455X (gfx1250) — matrix operand order repaired and checked by reading, not compiled, not run on hardware
//
#include <hip/hip_runtime.h>
#include <hip/hip_bf16.h>

// CausalSelfAttention for MI455X (gfx1250): bf16 WMMA + TDM async tile loads.
// B=4, T=2048, C=1024, H=16, hd=64.

#define N_HEAD   16
#define HEAD_DIM 64
#define EMB      1024
#define BATCH    4
#define SEQ      2048
#define BT       (BATCH * SEQ)          // 8192

// Fold 1/sqrt(hd) and log2(e) into Q so softmax can use exp2 (v_exp_f32).
#define QSCALE   (0.125f * 1.44269504088896340736f)

typedef __attribute__((ext_vector_type(16))) __bf16 v16bf;
typedef __attribute__((ext_vector_type(8)))  float  v8f;

union FragC { v8f v; float f[8]; };
union FragAB { v16bf v; unsigned u[8]; };

static __device__ inline v8f vzero8() {
  v8f z = {0.f, 0.f, 0.f, 0.f, 0.f, 0.f, 0.f, 0.f};
  return z;
}

static __device__ inline unsigned pack_bf16x2(float lo, float hi) {
  union { __bf16 h[2]; unsigned u; } p;
  p.h[0] = (__bf16)lo;
  p.h[1] = (__bf16)hi;
  return p.u;                      // v_cvt_pk_bf16_f32
}

static __device__ inline v8f wmma_bf16(v16bf a, v16bf b, v8f c) {
  // (neg_a, A, neg_b, B, c_mod, C, reuse_a, reuse_b)
  return __builtin_amdgcn_wmma_f32_16x16x32_bf16(false, a, false, b, (short)0, c,
                                                 false, false);
}

// A fragment (16x32 bf16, M x K), source [row][k] row-major, k contiguous.
// ISA: lane m = lane&15; lanes 0-15 hold K = 0..7,16..23; lanes 16-31 hold
// K = 8..15,24..31; each VGPR holds an even/odd K pair (dword-contiguous).
static __device__ inline v16bf load_frag_a(const __bf16* base, int ld, int row0, int k0) {
  const int lane = threadIdx.x & 31;
  const int m = lane & 15;
  const int kh = (lane >> 4) << 3;          // 0 or 8
  const __bf16* p = base + (size_t)(row0 + m) * ld + k0;
  FragAB f;
#pragma unroll
  for (int j = 0; j < 8; ++j) {
    const int k = (j < 4) ? (kh + 2 * j) : (16 + kh + 2 * (j - 4));
    f.u[j] = *(const unsigned*)(p + k);
  }
  return f.v;
}

// B fragment (32x16 bf16, K x N), source is B^T stored [n][k] row-major.
// ISA: lane n = lane&15; lanes 0-15 hold K = 0..7,16..23; lanes 16-31 hold K = 8..15,24..31.
static __device__ inline v16bf load_frag_b(const __bf16* base, int ld, int n0, int k0) {
  const int lane = threadIdx.x & 31;
  const int n = lane & 15;
  const int kh = (lane >> 4) << 3;          // 0 or 8
  const __bf16* p = base + (size_t)(n0 + n) * ld + k0;
  FragAB f;
#pragma unroll
  for (int j = 0; j < 8; ++j) f.u[j] = *(const unsigned*)(p + ((j < 4) ? (kh + 2 * j) : (16 + kh + 2 * (j - 4))));
  return f.v;
}

// ---------------------------------------------------------------------------
// TDM: async 64x64 bf16 tile load, global (row stride 64 elems) -> LDS with a
// 66-element (33-dword) row stride via the D# pad feature (pad_interval=32
// dwords, pad_amount=1 dword). Issued by one wave; completion via TENSORcnt.
// D# layout per CDNA5 ISA ch.8 (group0: count/lds/global/type, group1: dims).
// ---------------------------------------------------------------------------
typedef unsigned tdm_v4u __attribute__((ext_vector_type(4)));
typedef int      tdm_v8i __attribute__((ext_vector_type(8)));
typedef int      tdm_v4i __attribute__((ext_vector_type(4)));

static __device__ inline void tdm_load_tile64x64(const __bf16* gsrc,
                                                 const __bf16* lds_dst) {
  const unsigned long long ga = (unsigned long long)(size_t)(const void*)gsrc;
  const unsigned lds_addr = (unsigned)(size_t)(const void*)lds_dst;  // addr[31:0]

  tdm_v4u g0;
  g0.x = 0x1u;                                   // count=1 (valid user D#)
  g0.y = lds_addr;                               // lds_addr (bytes)
  g0.z = (unsigned)ga;                           // global_addr[31:0]
  g0.w = (unsigned)(ga >> 32) | (2u << 30);      // global_addr[56:32] | type=2

  tdm_v8i g1;
  g1[0] = (int)((1u << 16)        // data_size = 2 bytes
              | (1u << 20)        // pad_enable
              | (4u << 22));      // pad_interval = 32 dwords (64 bf16 per row)
                                  // pad_amount  = 1 dword  (code 0 in 31:25)
  g1[1] = (int)(64u << 16);       // tensor_dim0 = 64 (bits 79:48, low half)
  g1[2] = (int)(2048u << 16);     // tensor_dim1 = 2048 (bits 111:80, low half)
  g1[3] = (int)(64u << 16);       // tile_dim0 = 64 (bits 127:112)
  g1[4] = (int)64u;               // tile_dim1 = 64 rows (tile_dim2 = 0)
  g1[5] = (int)64u;               // tensor_dim0_stride = 64 elems (low 32)
  g1[6] = 0;                      // stride0 hi / tensor_dim1_stride lo
  g1[7] = 0;

  const tdm_v4i z4 = {0, 0, 0, 0};
#if __clang_major__ >= 23
  const tdm_v8i z8 = {0, 0, 0, 0, 0, 0, 0, 0};
  __builtin_amdgcn_tensor_load_to_lds(g0, g1, z4, z4, z8, 0);
#else
  __builtin_amdgcn_tensor_load_to_lds(g0, g1, z4, z4, 0);
#endif
}

// ---------------------------------------------------------------------------
// Shared GEMM body: 128(M) x 128(N) block tile, BK=64, 8 waves as 4(M) x 2(N),
// each wave 32x64 via 2 A-frags x 4 B-frags = 8 WMMA per 32-deep k-step.
// A and W are fp32 in global; converted to bf16 during LDS staging (packed).
// ---------------------------------------------------------------------------
static __device__ inline void gemm_body(const float* __restrict__ A,
                                        const float* __restrict__ W,
                                        int m0, int n0, int tid,
                                        __bf16* As, __bf16* Bs,
                                        v8f acc[2][4]) {
  const int wave = tid >> 5;
  const int wm = wave & 3;                  // 0..3 -> 32-row slice
  const int wn = wave >> 2;                 // 0..1 -> 64-col slice
  unsigned* Ad = (unsigned*)As;             // 33 dwords per row (66 bf16)
  unsigned* Bd = (unsigned*)Bs;

#pragma unroll 1
  for (int k0 = 0; k0 < EMB; k0 += 64) {
    // Stage A and B tiles (128x64 each): 4096 dwords / 256 threads = 16 each.
#pragma unroll
    for (int i = 0; i < 16; ++i) {
      const int idx = i * 256 + tid;        // 0..4095
      const int r = idx >> 5, cd = idx & 31;
      const float* sa = &A[(size_t)(m0 + r) * EMB + k0 + 2 * cd];
      const float* sb = &W[(size_t)(n0 + r) * EMB + k0 + 2 * cd];
      Ad[r * 33 + cd] = pack_bf16x2(sa[0], sa[1]);
      Bd[r * 33 + cd] = pack_bf16x2(sb[0], sb[1]);
    }
    if (k0 + 64 < EMB) {
      // Prefetch next K-tile (global_prefetch_b8).
      const int r = tid >> 5, cd = tid & 31;
      __builtin_prefetch(&A[(size_t)(m0 + r) * EMB + k0 + 64 + 2 * cd], 0, 1);
      __builtin_prefetch(&W[(size_t)(n0 + r) * EMB + k0 + 64 + 2 * cd], 0, 1);
    }
    __syncthreads();

#pragma unroll
    for (int ks = 0; ks < 2; ++ks) {
      const v16bf a0 = load_frag_a(As, 66, wm * 32, ks * 32);
      const v16bf a1 = load_frag_a(As, 66, wm * 32 + 16, ks * 32);
#pragma unroll
      for (int j = 0; j < 4; ++j) {
        const v16bf b = load_frag_b(Bs, 66, wn * 64 + j * 16, ks * 32);
        acc[0][j] = wmma_bf16(a0, b, acc[0][j]);
        acc[1][j] = wmma_bf16(a1, b, acc[1][j]);
      }
    }
    __syncthreads();
  }
}

// ---------------------------------------------------------------------------
// Kernel 1: qkv = x @ qkv_w^T + qkv_b, scattered to Q/K/V in [B,H,T,hd] bf16.
// Q is pre-scaled by QSCALE. The q/k/v selection is block-uniform (BN | 1024).
// ---------------------------------------------------------------------------
__global__ __launch_bounds__(256) void qkv_gemm_kernel(
    const float* __restrict__ x, const float* __restrict__ w,
    const float* __restrict__ bias,
    __bf16* __restrict__ Qb, __bf16* __restrict__ Kb, __bf16* __restrict__ Vb) {
  __shared__ __bf16 As[128 * 66];
  __shared__ __bf16 Bs[128 * 66];

  const int m0 = blockIdx.y * 128;
  const int n0 = blockIdx.x * 128;
  const int tid = threadIdx.x;
  const int lane = tid & 31;
  const int wave = tid >> 5;
  const int wm = wave & 3, wn = wave >> 2;

  v8f acc[2][4];
#pragma unroll
  for (int i = 0; i < 2; ++i)
#pragma unroll
    for (int j = 0; j < 4; ++j) acc[i][j] = vzero8();

  gemm_body(x, w, m0, n0, tid, As, Bs, acc);

  // Block-uniform destination select: 0=Q (scaled), 1=K, 2=V.
  const int which = n0 >> 10;
  __bf16* __restrict__ dst = (which == 0) ? Qb : (which == 1) ? Kb : Vb;
  const float scale = (which == 0) ? QSCALE : 1.0f;

  const int mrow_off = (lane >> 4) << 3;    // 0 or 8
  const int ncol = lane & 15;
  FragC cc;
#pragma unroll
  for (int i = 0; i < 2; ++i) {
#pragma unroll
    for (int j = 0; j < 4; ++j) {
      cc.v = acc[i][j];
#pragma unroll
      for (int r = 0; r < 8; ++r) {
        const int m = m0 + wm * 32 + i * 16 + mrow_off + r;   // [0,8192)
        const int n = n0 + wn * 64 + j * 16 + ncol;           // [0,3072)
        const float val = (cc.f[r] + bias[n]) * scale;
        const int cm = n & 1023;
        const int head = cm >> 6, d = cm & 63;
        const int bidx = m >> 11, t = m & 2047;
        dst[((size_t)(bidx * N_HEAD + head) * SEQ + t) * HEAD_DIM + d] = (__bf16)val;
      }
    }
  }
}

// ---------------------------------------------------------------------------
// Kernel 2: causal flash attention. Block = 128 queries of one (b,h);
// 8 waves x 16 query rows each; 64-key tiles. exp2-based online softmax.
// K tile arrives via TDM (async, overlapped with the V-transpose staging).
// ---------------------------------------------------------------------------
__global__ __launch_bounds__(256) void flash_attn_kernel(
    const __bf16* __restrict__ Qb, const __bf16* __restrict__ Kb,
    const __bf16* __restrict__ Vb, float* __restrict__ attn) {
  __shared__ __bf16 Ks[64 * 66];            // [key][hd]  (B^T layout for scores)
  __shared__ __bf16 Vs[64 * 66];            // [hd][key]  (B^T layout for P*V)
  __shared__ __bf16 Ps[8 * 16 * 66];        // per-wave P staging (C->A relayout)

  const int q0 = blockIdx.x * 128;
  const int bh = blockIdx.y;                // b*16 + h
  const int b = bh >> 4, h = bh & 15;
  const int tid = threadIdx.x, lane = tid & 31, wave = tid >> 5;
  const size_t base = (size_t)bh * SEQ * HEAD_DIM;
  const int qr = q0 + wave * 16;            // this wave's first query row

  // Q fragments straight from global (A layout; hd contiguous, dword pairs).
  v16bf qf[2];
  qf[0] = load_frag_a(Qb + base + (size_t)qr * HEAD_DIM, HEAD_DIM, 0, 0);
  qf[1] = load_frag_a(Qb + base + (size_t)qr * HEAD_DIM, HEAD_DIM, 0, 32);

  FragC o[4];
#pragma unroll
  for (int j = 0; j < 4; ++j) o[j].v = vzero8();
  float mrun[8], lrun[8];
#pragma unroll
  for (int r = 0; r < 8; ++r) { mrun[r] = -__builtin_inff(); lrun[r] = 0.f; }

  const int mrow_off = (lane >> 4) << 3;    // 0 or 8
  const int ncol = lane & 15;
  const int ntiles = (q0 >> 6) + 2;         // 64-key tiles up to q0+127
  __bf16* myP = Ps + wave * (16 * 66);

#pragma unroll 1
  for (int kt = 0; kt < ntiles; ++kt) {
    const int kbase = kt * 64;
    __syncthreads();                        // previous tile fully consumed

    // Wave 0 kicks off the async K-tile DMA (TENSORcnt-tracked) ...
    if (wave == 0)
      tdm_load_tile64x64(Kb + base + (size_t)kbase * HEAD_DIM, Ks);

    // ... while all waves stage V transposed [hd][key] (TDM can't transpose).
#pragma unroll
    for (int i = 0; i < 8; ++i) {
      const int idx = i * 256 + tid;        // 0..2047 dwords
      const int kk = idx >> 5, cd = idx & 31;
      union { unsigned u; __bf16 h[2]; } t;
      t.u = ((const unsigned*)(Vb + base + (size_t)(kbase + kk) * HEAD_DIM))[cd];
      Vs[(2 * cd) * 66 + kk] = t.h[0];
      Vs[(2 * cd + 1) * 66 + kk] = t.h[1];
    }
    if (wave == 0) __builtin_amdgcn_s_wait_tensorcnt(0);
    __syncthreads();                        // K tile + V tile visible to all

    // Scores: 16 queries x 64 keys (4 N-frags), K-depth 64 (2 k-steps) = 8 WMMA.
    FragC s[4];
#pragma unroll
    for (int j = 0; j < 4; ++j) {
      s[j].v = vzero8();
      s[j].v = wmma_bf16(qf[0], load_frag_b(Ks, 66, j * 16, 0), s[j].v);
      s[j].v = wmma_bf16(qf[1], load_frag_b(Ks, 66, j * 16, 32), s[j].v);
    }

    // Causal mask + online softmax (base-2). Row m spans 16 lanes (half-wave).
#pragma unroll
    for (int r = 0; r < 8; ++r) {
      const int m = qr + mrow_off + r;
      float t = -__builtin_inff();
#pragma unroll
      for (int j = 0; j < 4; ++j) {
        s[j].f[r] = ((kbase + j * 16 + ncol) > m) ? -__builtin_inff() : s[j].f[r];
        t = fmaxf(t, s[j].f[r]);
      }
#pragma unroll
      for (int off = 8; off; off >>= 1) t = fmaxf(t, __shfl_xor(t, off, 16));
      const float mn = fmaxf(mrun[r], t);   // finite: key 0 always visible
      const float corr = exp2f(mrun[r] - mn);
      mrun[r] = mn;
      float rs = 0.f;
      const int ml = mrow_off + r;
#pragma unroll
      for (int j = 0; j < 4; ++j) {
        const float p = exp2f(s[j].f[r] - mn);
        rs += p;
        myP[ml * 66 + j * 16 + ncol] = (__bf16)p;   // C-layout -> A-layout
      }
#pragma unroll
      for (int off = 8; off; off >>= 1) rs += __shfl_xor(rs, off, 16);
      lrun[r] = lrun[r] * corr + rs;
#pragma unroll
      for (int j = 0; j < 4; ++j) o[j].f[r] *= corr;
    }

    // P (16x64) @ V (64x64): 2 A-frags x 4 N-frags = 8 WMMA into O.
    const v16bf pa0 = load_frag_a(myP, 66, 0, 0);
    const v16bf pa1 = load_frag_a(myP, 66, 0, 32);
#pragma unroll
    for (int j = 0; j < 4; ++j) {
      o[j].v = wmma_bf16(pa0, load_frag_b(Vs, 66, j * 16, 0), o[j].v);
      o[j].v = wmma_bf16(pa1, load_frag_b(Vs, 66, j * 16, 32), o[j].v);
    }
  }

  // Epilogue: normalize, write to attn [B,T,C] fp32.
#pragma unroll
  for (int j = 0; j < 4; ++j) {
#pragma unroll
    for (int r = 0; r < 8; ++r) {
      const int t = qr + mrow_off + r;
      const int c = h * HEAD_DIM + j * 16 + ncol;
      attn[((size_t)(b * SEQ + t)) * EMB + c] = o[j].f[r] / lrun[r];
    }
  }
}

// ---------------------------------------------------------------------------
// Kernel 3: out = attn @ proj_w^T + proj_b  (fp32 out). Same tiling as K1.
// ---------------------------------------------------------------------------
__global__ __launch_bounds__(256) void proj_gemm_kernel(
    const float* __restrict__ a, const float* __restrict__ w,
    const float* __restrict__ bias, float* __restrict__ out) {
  __shared__ __bf16 As[128 * 66];
  __shared__ __bf16 Bs[128 * 66];

  const int m0 = blockIdx.y * 128;
  const int n0 = blockIdx.x * 128;
  const int tid = threadIdx.x;
  const int lane = tid & 31;
  const int wave = tid >> 5;
  const int wm = wave & 3, wn = wave >> 2;

  v8f acc[2][4];
#pragma unroll
  for (int i = 0; i < 2; ++i)
#pragma unroll
    for (int j = 0; j < 4; ++j) acc[i][j] = vzero8();

  gemm_body(a, w, m0, n0, tid, As, Bs, acc);

  const int mrow_off = (lane >> 4) << 3;
  const int ncol = lane & 15;
  FragC cc;
#pragma unroll
  for (int i = 0; i < 2; ++i) {
#pragma unroll
    for (int j = 0; j < 4; ++j) {
      cc.v = acc[i][j];
#pragma unroll
      for (int r = 0; r < 8; ++r) {
        const int m = m0 + wm * 32 + i * 16 + mrow_off + r;
        const int n = n0 + wn * 64 + j * 16 + ncol;
        out[(size_t)m * EMB + n] = cc.f[r] + bias[n];
      }
    }
  }
}

// ---------------------------------------------------------------------------
extern "C" void kernel_launch(void* const* d_in, const int* in_sizes, int n_in,
                              void* d_out, int out_size, void* d_ws, size_t ws_size,
                              hipStream_t stream) {
  (void)in_sizes; (void)n_in; (void)out_size; (void)ws_size;
  const float* x      = (const float*)d_in[0];
  const float* qkv_w  = (const float*)d_in[1];
  const float* qkv_b  = (const float*)d_in[2];
  const float* proj_w = (const float*)d_in[3];
  const float* proj_b = (const float*)d_in[4];
  float* out = (float*)d_out;

  // Workspace carve-up (80 MB total):
  //   Qb/Kb/Vb: bf16 [B,H,T,hd] = 16 MB each; attn: fp32 [B,T,C] = 32 MB.
  const size_t NELEM = (size_t)BATCH * N_HEAD * SEQ * HEAD_DIM;   // 8,388,608
  char* ws = (char*)d_ws;
  __bf16* Qb = (__bf16*)ws;
  __bf16* Kb = Qb + NELEM;
  __bf16* Vb = Kb + NELEM;
  float* attn = (float*)(ws + 3 * NELEM * sizeof(__bf16));

  // 1) QKV projection: [8192,1024] x [3072,1024]^T
  qkv_gemm_kernel<<<dim3(3 * EMB / 128, BT / 128), 256, 0, stream>>>(
      x, qkv_w, qkv_b, Qb, Kb, Vb);

  // 2) Causal flash attention: grid = (T/128, B*H)
  flash_attn_kernel<<<dim3(SEQ / 128, BATCH * N_HEAD), 256, 0, stream>>>(
      Qb, Kb, Vb, attn);

  // 3) Output projection: [8192,1024] x [1024,1024]^T
  proj_gemm_kernel<<<dim3(EMB / 128, BT / 128), 256, 0, stream>>>(
      attn, proj_w, proj_b, out);
}
